// AttentionHead_28802050687381
// MI455X (gfx1250) — compile-verified, matrix operand order checked by reading, not run on hardware
//
#include <hip/hip_runtime.h>
#include <hip/hip_bf16.h>

// ---------------------------------------------------------------------------
// MI455X (gfx1250) causal MHA block.
// All matmuls via v_wmma_f32_16x16x32_bf16 (wave32 WMMA, f32 accum).
// ~50MB working set << 192MB L2 -> compute bound -> bf16 WMMA everywhere.
// Flash attention streams 32-key tiles through double-buffered LDS using
// GLOBAL_LOAD_ASYNC_TO_LDS_B128 (ASYNCcnt-tracked) when the toolchain
// exposes it. V is stored pre-transposed [B,H,D,T] by the QKV epilogue so
// all tile staging is pure 16B-chunk copies (no in-kernel transpose).
// ---------------------------------------------------------------------------

typedef __bf16 bf16_t;
typedef __attribute__((ext_vector_type(4)))  __bf16 v4bf;
typedef __attribute__((ext_vector_type(8)))  __bf16 v8bf;
typedef __attribute__((ext_vector_type(16))) __bf16 v16bf;
typedef __attribute__((ext_vector_type(8)))  float  v8f;
typedef __attribute__((ext_vector_type(4)))  int    v4i;

// pointee typedefs carrying the address space (attribute attached to the
// type itself so C-style casts below perform reinterpret + addrspacecast)
typedef __attribute__((address_space(1))) v4i v4i_global;
typedef __attribute__((address_space(3))) v4i v4i_lds;

#define BDIM 2
#define TDIM 2048
#define CDIM 1024
#define HDIM 16
#define DDIM 64
#define LOG2E 1.44269504088896340736f

#if __has_builtin(__builtin_amdgcn_global_load_async_to_lds_b128) && \
    __has_builtin(__builtin_amdgcn_s_wait_asynccnt)
#define HAVE_ASYNC_LDS 1
#else
#define HAVE_ASYNC_LDS 0
#endif

__device__ __forceinline__ v8f vzero8() {
  v8f v;
#pragma unroll
  for (int i = 0; i < 8; ++i) v[i] = 0.0f;
  return v;
}

// Copy one 16-byte chunk global -> LDS. Async (ASYNCcnt-tracked, no VGPR data
// path) when available, else plain b128 load + ds_store.
__device__ __forceinline__ void cp16(bf16_t* lds_dst, const bf16_t* gsrc) {
#if HAVE_ASYNC_LDS
  __builtin_amdgcn_global_load_async_to_lds_b128(
      (v4i_global*)gsrc, (v4i_lds*)lds_dst, 0, 0);
#else
  *(v8bf*)lds_dst = *(const v8bf*)gsrc;
#endif
}

__device__ __forceinline__ void wait_async_copies() {
#if HAVE_ASYNC_LDS
  __builtin_amdgcn_s_wait_asynccnt(0);
#endif
}

// Load one 16-element bf16 A/B fragment row-chunk: elements 0..7 live at p,
// elements 8..15 at p+16 (K jumps by 16 between the two VGPR quads).
// Both halves are contiguous 16-byte chunks -> two b128 loads.
__device__ __forceinline__ v16bf load_frag(const bf16_t* p) {
  v8bf lo = *(const v8bf*)(p);
  v8bf hi = *(const v8bf*)(p + 16);
  return __builtin_shufflevector(lo, hi, 0, 1, 2, 3, 4, 5, 6, 7,
                                 8, 9, 10, 11, 12, 13, 14, 15);
}

// A (16xK, row=M) or B (KxN stored N-major, row=N) fragment gather from a
// row-major [row][ld] bf16 array, per CDNA5 ISA 7.12.2 16-bit layout:
//   lane L -> row = row0 + (L&15);  K = kbase + (L>>4)*8 + {0..7, 16..23}
__device__ __forceinline__ v16bf frag_rm(const bf16_t* base, int row0, int ld,
                                         int kbase, int lane) {
  const bf16_t* p =
      base + (size_t)(row0 + (lane & 15)) * ld + kbase + ((lane >> 4) << 3);
  return load_frag(p);
}

__device__ __forceinline__ v8f wmma_bf16(v16bf a, v16bf b, v8f c) {
  return __builtin_amdgcn_wmma_f32_16x16x32_bf16(false, a, false, b, (short)0,
                                                 c, false, false);
}

// ---------------------------------------------------------------------------
// f32 -> bf16 conversion (float4 in, 8-byte bf16x4 out)
// ---------------------------------------------------------------------------
__global__ __launch_bounds__(256) void cvt_bf16_kernel(
    const float* __restrict__ src, bf16_t* __restrict__ dst, int n4) {
  int i = blockIdx.x * blockDim.x + threadIdx.x;
  if (i < n4) {
    float4 f = ((const float4*)src)[i];
    v4bf o;
    o[0] = (bf16_t)f.x;
    o[1] = (bf16_t)f.y;
    o[2] = (bf16_t)f.z;
    o[3] = (bf16_t)f.w;
    ((v4bf*)dst)[i] = o;
  }
}

// ---------------------------------------------------------------------------
// QKV GEMM: qkv[m,n] = X[m,:] . Wqkv[n,:] + b[n], scattered to per-head
// tensors: Q,K as [B,H,T,D]; V pre-TRANSPOSED as [B,H,D,T] for the flash
// kernel's B-fragments. 1/sqrt(C) folded into Q.
// Block: 256 thr (8 waves), tile 128(M)x128(N); wave tile 32x64.
// ---------------------------------------------------------------------------
__global__ __launch_bounds__(256) void qkv_gemm_kernel(
    const bf16_t* __restrict__ X,      // [B*T, C]
    const bf16_t* __restrict__ W,      // [3C, C]
    const float* __restrict__ bias,    // [3C]
    bf16_t* __restrict__ Qo, bf16_t* __restrict__ Ko,
    bf16_t* __restrict__ Vto) {
  const int lane = threadIdx.x & 31;
  const int wave = threadIdx.x >> 5;
  const int wm = wave & 3, wn = wave >> 2;
  const int m0 = blockIdx.x * 128 + wm * 32;
  const int n0 = blockIdx.y * 128 + wn * 64;

  v8f acc[2][4];
#pragma unroll
  for (int i = 0; i < 2; ++i)
#pragma unroll
    for (int j = 0; j < 4; ++j) acc[i][j] = vzero8();

  for (int kb = 0; kb < CDIM; kb += 32) {
    v16bf a0 = frag_rm(X, m0, CDIM, kb, lane);
    v16bf a1 = frag_rm(X, m0 + 16, CDIM, kb, lane);
    v16bf b0 = frag_rm(W, n0, CDIM, kb, lane);
    v16bf b1 = frag_rm(W, n0 + 16, CDIM, kb, lane);
    v16bf b2 = frag_rm(W, n0 + 32, CDIM, kb, lane);
    v16bf b3 = frag_rm(W, n0 + 48, CDIM, kb, lane);
    acc[0][0] = wmma_bf16(a0, b0, acc[0][0]);
    acc[0][1] = wmma_bf16(a0, b1, acc[0][1]);
    acc[0][2] = wmma_bf16(a0, b2, acc[0][2]);
    acc[0][3] = wmma_bf16(a0, b3, acc[0][3]);
    acc[1][0] = wmma_bf16(a1, b0, acc[1][0]);
    acc[1][1] = wmma_bf16(a1, b1, acc[1][1]);
    acc[1][2] = wmma_bf16(a1, b2, acc[1][2]);
    acc[1][3] = wmma_bf16(a1, b3, acc[1][3]);
  }

  // C/D layout: element r of lane -> M = r + 8*(lane>>4), N = lane&15
  const int mh = (lane >> 4) << 3;
  const int nl = lane & 15;
#pragma unroll
  for (int j = 0; j < 4; ++j) {
    const int gn = n0 + j * 16 + nl;  // 0..3C
    const int sel = gn >> 10;         // 0=Q 1=K 2=V
    const int oc = gn & (CDIM - 1);
    const int h = oc >> 6, d = oc & 63;
    const float bs = bias[gn];
    const float sc = (sel == 0) ? 0.03125f : 1.0f;  // fold 1/sqrt(C) into Q
#pragma unroll
    for (int i = 0; i < 2; ++i) {
#pragma unroll
      for (int r = 0; r < 8; ++r) {
        const int gm = m0 + i * 16 + mh + r;  // = b*T + t
        const int bb = gm >> 11;
        const int t = gm & (TDIM - 1);
        const bf16_t val = (bf16_t)((acc[i][j][r] + bs) * sc);
        if (sel == 2) {
          // V transposed: [B,H,D,T]
          Vto[((size_t)((bb * HDIM + h) * DDIM + d)) * TDIM + t] = val;
        } else {
          bf16_t* dst = (sel == 0) ? Qo : Ko;
          dst[((size_t)((bb * HDIM + h) * TDIM + t)) * DDIM + d] = val;
        }
      }
    }
  }
}

// ---------------------------------------------------------------------------
// Flash attention: one block = 64 query rows of one (b,h). 4 waves, each owns
// 16 query rows. Streams 32-key tiles through DOUBLE-BUFFERED LDS via async
// global->LDS copies (overlapped with WMMA + softmax of the current tile).
// Never materializes T x T.
//   kt buf: K tile   [s=32][d=64]   row-major (B-frags for Q.K^T)
//   vt buf: V^T tile [d=64][s=32]   stride 40 (B-frags for P.V)
// ---------------------------------------------------------------------------
__global__ __launch_bounds__(128) void flash_attn_kernel(
    const bf16_t* __restrict__ Q, const bf16_t* __restrict__ K,
    const bf16_t* __restrict__ Vt, bf16_t* __restrict__ Y) {
  __shared__ __align__(16) bf16_t kt[2][32 * 64];
  __shared__ __align__(16) bf16_t vt[2][64 * 40];
  __shared__ __align__(16) bf16_t pt[4][16 * 40];

  const int tid = threadIdx.x;
  const int lane = tid & 31;
  const int wave = tid >> 5;   // 0..3
  const int bh = blockIdx.y;   // b*H + h
  const int q0 = blockIdx.x * 64;
  const int qw = q0 + wave * 16;

  const bf16_t* Qp = Q + (size_t)bh * TDIM * DDIM;
  const bf16_t* Kp = K + (size_t)bh * TDIM * DDIM;
  const bf16_t* Vtp = Vt + (size_t)bh * DDIM * TDIM;  // [64][2048]

  // Q fragments for this wave's 16 rows (live in VGPRs for the whole kernel)
  v16bf qf0 = frag_rm(Qp, qw, DDIM, 0, lane);
  v16bf qf1 = frag_rm(Qp, qw, DDIM, 32, lane);

  v8f o[4];
#pragma unroll
  for (int j = 0; j < 4; ++j) o[j] = vzero8();
  float mrow[8], lrow[8];
#pragma unroll
  for (int r = 0; r < 8; ++r) {
    mrow[r] = -__builtin_inff();
    lrow[r] = 0.0f;
  }

  const int mh = (lane >> 4) << 3;
  const int nl = lane & 15;

  // issue K tile (256 chunks) + V^T tile (64 rows x 4 chunks): 4 cp16/thread
  auto load_tiles = [&](int s0, bf16_t* ktb, bf16_t* vtb) {
#pragma unroll
    for (int c = tid; c < 256; c += 128)
      cp16(ktb + c * 8, Kp + (size_t)s0 * DDIM + c * 8);
#pragma unroll
    for (int c = tid; c < 256; c += 128) {
      const int d = c >> 2, pcol = (c & 3) * 8;
      cp16(vtb + d * 40 + pcol, Vtp + (size_t)d * TDIM + s0 + pcol);
    }
  };

  const int nIter = q0 / 32 + 2;  // keys 0 .. q0+63 in 32-wide tiles
  load_tiles(0, kt[0], vt[0]);

  for (int it = 0; it < nIter; ++it) {
    const int s0 = it * 32;
    const int cur = it & 1;
    wait_async_copies();   // own async chunks done
    __syncthreads();       // everyone's chunks done; prev buf fully consumed
    if (it + 1 < nIter)    // prefetch next tile, overlapped with compute
      load_tiles(s0 + 32, kt[cur ^ 1], vt[cur ^ 1]);

    const bf16_t* ktc = kt[cur];
    const bf16_t* vtc = vt[cur];

    // scores S = Q.K^T (pre-scaled): 16 x 32, two 16x16 accumulators
    v8f sa0 = vzero8(), sa1 = vzero8();
    sa0 = wmma_bf16(qf0, frag_rm(ktc, 0, 64, 0, lane), sa0);
    sa1 = wmma_bf16(qf0, frag_rm(ktc, 16, 64, 0, lane), sa1);
    sa0 = wmma_bf16(qf1, frag_rm(ktc, 0, 64, 32, lane), sa0);
    sa1 = wmma_bf16(qf1, frag_rm(ktc, 16, 64, 32, lane), sa1);

    // causal mask + online softmax (per query row)
#pragma unroll
    for (int r = 0; r < 8; ++r) {
      const int qrow = qw + mh + r;
      float s0v = sa0[r];
      float s1v = sa1[r];
      if (s0 + nl > qrow) s0v = -__builtin_inff();
      if (s0 + 16 + nl > qrow) s1v = -__builtin_inff();
      float mx = fmaxf(s0v, s1v);
      mx = fmaxf(mx, __shfl_xor(mx, 1, 32));
      mx = fmaxf(mx, __shfl_xor(mx, 2, 32));
      mx = fmaxf(mx, __shfl_xor(mx, 4, 32));
      mx = fmaxf(mx, __shfl_xor(mx, 8, 32));
      const float mnew = fmaxf(mrow[r], mx);
      const float alpha = exp2f((mrow[r] - mnew) * LOG2E);
      const float p0 = exp2f((s0v - mnew) * LOG2E);
      const float p1 = exp2f((s1v - mnew) * LOG2E);
      float ps = p0 + p1;
      ps += __shfl_xor(ps, 1, 32);
      ps += __shfl_xor(ps, 2, 32);
      ps += __shfl_xor(ps, 4, 32);
      ps += __shfl_xor(ps, 8, 32);
      lrow[r] = lrow[r] * alpha + ps;
      mrow[r] = mnew;
#pragma unroll
      for (int j = 0; j < 4; ++j) o[j][r] *= alpha;
      // C-layout -> LDS (re-layout into A-fragment order for P@V)
      pt[wave][(mh + r) * 40 + nl] = (bf16_t)p0;
      pt[wave][(mh + r) * 40 + 16 + nl] = (bf16_t)p1;
    }

    // O += P @ V  (A = 16x32 probs, B = V^T tile, 4 WMMA over D)
    v16bf pf = frag_rm(pt[wave], 0, 40, 0, lane);
    o[0] = wmma_bf16(pf, frag_rm(vtc, 0, 40, 0, lane), o[0]);
    o[1] = wmma_bf16(pf, frag_rm(vtc, 16, 40, 0, lane), o[1]);
    o[2] = wmma_bf16(pf, frag_rm(vtc, 32, 40, 0, lane), o[2]);
    o[3] = wmma_bf16(pf, frag_rm(vtc, 48, 40, 0, lane), o[3]);
  }

  // epilogue: O /= l, write bf16 y[B,T,C] (heads re-interleaved)
  const int bb = bh >> 4;
  const int h = bh & 15;
#pragma unroll
  for (int r = 0; r < 8; ++r) {
    const float inv = 1.0f / lrow[r];
    const int t = qw + mh + r;
#pragma unroll
    for (int j = 0; j < 4; ++j)
      Y[((size_t)(bb * TDIM + t)) * CDIM + h * DDIM + j * 16 + nl] =
          (bf16_t)(o[j][r] * inv);
  }
}

// ---------------------------------------------------------------------------
// Output projection: out[m,n] = Y[m,:] . Wproj[n,:] + b[n]  (f32 out)
// ---------------------------------------------------------------------------
__global__ __launch_bounds__(256) void proj_gemm_kernel(
    const bf16_t* __restrict__ Yb,    // [B*T, C]
    const bf16_t* __restrict__ W,     // [C, C]
    const float* __restrict__ bias,   // [C]
    float* __restrict__ Out) {        // [B*T, C]
  const int lane = threadIdx.x & 31;
  const int wave = threadIdx.x >> 5;
  const int wm = wave & 3, wn = wave >> 2;
  const int m0 = blockIdx.x * 128 + wm * 32;
  const int n0 = blockIdx.y * 128 + wn * 64;

  v8f acc[2][4];
#pragma unroll
  for (int i = 0; i < 2; ++i)
#pragma unroll
    for (int j = 0; j < 4; ++j) acc[i][j] = vzero8();

  for (int kb = 0; kb < CDIM; kb += 32) {
    v16bf a0 = frag_rm(Yb, m0, CDIM, kb, lane);
    v16bf a1 = frag_rm(Yb, m0 + 16, CDIM, kb, lane);
    v16bf b0 = frag_rm(W, n0, CDIM, kb, lane);
    v16bf b1 = frag_rm(W, n0 + 16, CDIM, kb, lane);
    v16bf b2 = frag_rm(W, n0 + 32, CDIM, kb, lane);
    v16bf b3 = frag_rm(W, n0 + 48, CDIM, kb, lane);
    acc[0][0] = wmma_bf16(a0, b0, acc[0][0]);
    acc[0][1] = wmma_bf16(a0, b1, acc[0][1]);
    acc[0][2] = wmma_bf16(a0, b2, acc[0][2]);
    acc[0][3] = wmma_bf16(a0, b3, acc[0][3]);
    acc[1][0] = wmma_bf16(a1, b0, acc[1][0]);
    acc[1][1] = wmma_bf16(a1, b1, acc[1][1]);
    acc[1][2] = wmma_bf16(a1, b2, acc[1][2]);
    acc[1][3] = wmma_bf16(a1, b3, acc[1][3]);
  }

  const int mh = (lane >> 4) << 3;
  const int nl = lane & 15;
#pragma unroll
  for (int j = 0; j < 4; ++j) {
    const int gn = n0 + j * 16 + nl;
    const float bs = bias[gn];
#pragma unroll
    for (int i = 0; i < 2; ++i) {
#pragma unroll
      for (int r = 0; r < 8; ++r) {
        const int gm = m0 + i * 16 + mh + r;
        Out[(size_t)gm * CDIM + gn] = acc[i][j][r] + bs;
      }
    }
  }
}

// ---------------------------------------------------------------------------
extern "C" void kernel_launch(void* const* d_in, const int* in_sizes, int n_in,
                              void* d_out, int out_size, void* d_ws,
                              size_t ws_size, hipStream_t stream) {
  const float* x = (const float*)d_in[0];       // [B,T,C]
  const float* w_qkv = (const float*)d_in[1];   // [3C,C]
  const float* b_qkv = (const float*)d_in[2];   // [3C]
  const float* w_proj = (const float*)d_in[3];  // [C,C]
  const float* b_proj = (const float*)d_in[4];  // [C]
  float* out = (float*)d_out;                   // [B,T,C] f32

  const size_t nx = (size_t)BDIM * TDIM * CDIM;
  const size_t nwq = (size_t)3 * CDIM * CDIM;
  const size_t nwp = (size_t)CDIM * CDIM;
  const size_t nh = (size_t)BDIM * HDIM * TDIM * DDIM;

  bf16_t* p = (bf16_t*)d_ws;
  bf16_t* xb = p;      p += nx;
  bf16_t* wqkvb = p;   p += nwq;
  bf16_t* wprojb = p;  p += nwp;
  bf16_t* Qb = p;      p += nh;
  bf16_t* Kb = p;      p += nh;
  bf16_t* Vtb = p;     p += nh;   // transposed [B,H,D,T]
  bf16_t* yb = p;      p += nx;

  // 1) f32 -> bf16 conversions
  cvt_bf16_kernel<<<(int)(nx / 4 / 256), 256, 0, stream>>>(x, xb,
                                                           (int)(nx / 4));
  cvt_bf16_kernel<<<(int)(nwq / 4 / 256), 256, 0, stream>>>(w_qkv, wqkvb,
                                                            (int)(nwq / 4));
  cvt_bf16_kernel<<<(int)(nwp / 4 / 256), 256, 0, stream>>>(w_proj, wprojb,
                                                            (int)(nwp / 4));

  // 2) QKV projection + head scatter (M=4096, N=3072, K=1024)
  qkv_gemm_kernel<<<dim3(4096 / 128, 3072 / 128), 256, 0, stream>>>(
      xb, wqkvb, b_qkv, Qb, Kb, Vtb);

  // 3) flash attention: grid = (T/64 query tiles, B*H)
  flash_attn_kernel<<<dim3(TDIM / 64, BDIM * HDIM), 128, 0, stream>>>(Qb, Kb,
                                                                      Vtb, yb);

  // 4) output projection (M=4096, N=1024, K=1024), f32 epilogue
  proj_gemm_kernel<<<dim3(4096 / 128, 1024 / 128), 256, 0, stream>>>(
      yb, wprojb, b_proj, out);
}
